// DynamicModel_9088150798672
// MI455X (gfx1250) — matrix operand order repaired and checked by reading, not compiled, not run on hardware
//
#include <hip/hip_runtime.h>

typedef __attribute__((ext_vector_type(16))) _Float16 v16h;
typedef __attribute__((ext_vector_type(8)))  float    v8f;

#define T_LEN 1024
#define HDIM  512
#define CMCH  8     // 4 input channels + 4 output channels
#define NB    16    // batch tile = WMMA N
#define COUT  4
#define CIN   4

// One workgroup = 16 waves (512 threads), handles NB=16 batch elements for all
// T=1024 recurrent steps. Wave w owns H rows [32w, 32w+32) of W1, held in
// registers as f16 WMMA A-tiles. Signal history (u and fed-back y) lives in a
// 64-deep f16 ring buffer in LDS; each step it is restaged into the WMMA
// B-matrix lane layout, 16 v_wmma_f32_16x16x32_f16 per wave accumulate
// h = W1flat @ x_t (+b1 via accumulator init), relu, then W2@h is reduced with
// per-lane FMAs + ds_add_f32 atomics, and y(t) is written to global + ring.
__global__ __launch_bounds__(512, 1)
void recur_wmma_kernel(const float* __restrict__ u,
                       const float* __restrict__ W1,
                       const float* __restrict__ b1,
                       const float* __restrict__ W2,
                       const float* __restrict__ b2,
                       float* __restrict__ out)
{
    __shared__ __align__(32) _Float16 ring[CMCH][NB][64];    // 16 KB: S history, f16
    __shared__ __align__(32) _Float16 staged[CMCH][32][16];  //  8 KB: B tiles in WMMA layout
    __shared__ float ybuf[COUT * NB];                        // 256 B: y reduction

    const int tid  = threadIdx.x;
    const int lane = tid & 31;
    const int w    = tid >> 5;      // wave id 0..15 -> H rows [32w, 32w+32)
    const int g    = lane >> 4;     // lane group
    const int n    = lane & 15;     // WMMA N index = batch within tile (also M for A load)
    const int n0   = blockIdx.x * NB;

    // ---- zero ring (history times < 0 are zero) and ybuf ----
    for (int i = tid; i < CMCH * NB * 64; i += 512)
        ((_Float16*)ring)[i] = (_Float16)0.f;
    if (tid < COUT * NB) ybuf[tid] = 0.f;

    // ---- preload W1 rows as f16 A-tiles: a[mi][kt], M = lane%16, K per ISA layout ----
    v16h a[2][8];
#pragma unroll
    for (int mi = 0; mi < 2; ++mi) {
        const int row = w * 32 + mi * 16 + n;   // global H row (A-matrix M = lane%16)
#pragma unroll
        for (int kt = 0; kt < 8; ++kt) {
#pragma unroll
            for (int h = 0; h < 16; ++h) {
                const int K = ((h < 8) ? h : h + 8) + 8 * g;  // A-matrix 16-bit layout
                a[mi][kt][h] = (_Float16)W1[row * 256 + kt * 32 + K];
            }
        }
    }

    // ---- bias as pre-initialized accumulators (C/D layout: M = r + 8g + 16mi + 32w) ----
    v8f biasacc[2];
#pragma unroll
    for (int mi = 0; mi < 2; ++mi)
#pragma unroll
        for (int r = 0; r < 8; ++r)
            biasacc[mi][r] = b1[w * 32 + mi * 16 + g * 8 + r];

    // ---- W2 columns for this lane's 16 h-values ----
    float w2r[COUT][2][8];
#pragma unroll
    for (int o = 0; o < COUT; ++o)
#pragma unroll
        for (int mi = 0; mi < 2; ++mi)
#pragma unroll
            for (int r = 0; r < 8; ++r)
                w2r[o][mi][r] = W2[o * HDIM + w * 32 + mi * 16 + g * 8 + r];

    const float b2c = b2[tid & 3];  // unused: the y-writeback reads b2[o]

    // staging work decomposition: 4096 staged halves / 512 threads = 8 each
    const int s_kt = tid >> 6;          // tile (= channel) 0..7
    const int s_L  = (tid >> 1) & 31;   // lane slot in staged tile
    const int s_g  = s_L >> 4;
    const int s_n  = s_L & 15;
    const int s_h0 = (tid & 1) * 8;

    __syncthreads();

    for (int t = 0; t < T_LEN; ++t) {
        // ---- stage B tiles from ring: B layout N=lane%16, K = ((h<8)?h:h+8) + 8*(lane/16),
        //      value = S[channel kt][batch n][time t-32+K] ----
        {
            const int tb = t - 32 + 8 * s_g + 2 * s_h0;
#pragma unroll
            for (int h = 0; h < 8; ++h)
                staged[s_kt][s_L][s_h0 + h] = ring[s_kt][s_n][(tb + h) & 63];
        }
        __syncthreads();

        // ---- h = relu(W1flat @ x_t + b1): 16 WMMAs per wave ----
        v8f acc0 = biasacc[0];
        v8f acc1 = biasacc[1];
#pragma unroll
        for (int kt = 0; kt < 8; ++kt) {
            const v16h b = *(const v16h*)&staged[kt][lane][0];
            acc0 = __builtin_amdgcn_wmma_f32_16x16x32_f16(
                       false, a[0][kt], false, b, (short)0, acc0, false, false);
            acc1 = __builtin_amdgcn_wmma_f32_16x16x32_f16(
                       false, a[1][kt], false, b, (short)0, acc1, false, false);
        }

        // ---- relu + per-lane W2 partials, reduce via LDS float atomics ----
        float p[COUT] = {0.f, 0.f, 0.f, 0.f};
#pragma unroll
        for (int r = 0; r < 8; ++r) {
            const float h0v = fmaxf(acc0[r], 0.f);
            const float h1v = fmaxf(acc1[r], 0.f);
#pragma unroll
            for (int o = 0; o < COUT; ++o) {
                p[o] = fmaf(w2r[o][0][r], h0v, p[o]);
                p[o] = fmaf(w2r[o][1][r], h1v, p[o]);
            }
        }
#pragma unroll
        for (int o = 0; o < COUT; ++o)
            atomicAdd(&ybuf[o * NB + n], p[o]);
        __syncthreads();

        // ---- finalize y(t): global store + feed back into ring; refill u(t) ----
        if (tid < COUT * NB) {                 // 64 threads: o = tid>>4, nn = tid&15
            const int o  = tid >> 4;
            const int nn = tid & 15;
            const float yv = ybuf[tid] + b2[o];
            out[((size_t)(n0 + nn) * COUT + o) * T_LEN + t] = yv;
            ring[CIN + o][nn][t & 63] = (_Float16)yv;
            ybuf[tid] = 0.f;
        } else if (tid < 2 * COUT * NB) {      // 64 threads: write u(t) into ring
            const int idx = tid - COUT * NB;
            const int c   = idx >> 4;
            const int nn  = idx & 15;
            ring[c][nn][t & 63] =
                (_Float16)u[((size_t)(n0 + nn) * CIN + c) * T_LEN + t];
        }
        (void)b2c;
        __syncthreads();
    }
}

extern "C" void kernel_launch(void* const* d_in, const int* in_sizes, int n_in,
                              void* d_out, int out_size, void* d_ws, size_t ws_size,
                              hipStream_t stream)
{
    const float* u  = (const float*)d_in[0];
    const float* W1 = (const float*)d_in[1];
    const float* b1 = (const float*)d_in[2];
    const float* W2 = (const float*)d_in[3];
    const float* b2 = (const float*)d_in[4];
    float* out = (float*)d_out;

    const int B = in_sizes[0] / (CIN * T_LEN);   // 64
    const int nblocks = B / NB;                  // 4 workgroups of 16 waves

    recur_wmma_kernel<<<dim3(nblocks), dim3(512), 0, stream>>>(u, W1, b1, W2, b2, out);
}
